// AttentionModule_1297080123532
// MI455X (gfx1250) — matrix operand order repaired and checked by reading, not compiled, not run on hardware
//
#include <hip/hip_runtime.h>
#include <hip/hip_bf16.h>
#include <cstdint>

// ---------------------------------------------------------------------------
// CDNA5 (gfx1250) bf16 WMMA attention pipeline, v3.1.
//  - Both projection GEMMs are LDS-free: A/B fragments stream straight from
//    global memory as 128-bit loads feeding v_wmma_f32_16x16x32_bf16.
//  - x transpose tile staged via TDM tensor_load_to_lds (TENSORcnt) when the
//    builtin exists (6-arg clang-23 form), else async global->LDS b128.
//  - Attention: each wave owns 2 whole 16x16 score tiles over full K
//    (no cross-wave reduction), fp32 softmax, transposed packed output.
// ---------------------------------------------------------------------------

typedef __attribute__((ext_vector_type(16))) __bf16 v16bf;
typedef __attribute__((ext_vector_type(8)))  float  v8f;
typedef __attribute__((ext_vector_type(4)))  unsigned int u32x4;
typedef __attribute__((ext_vector_type(8)))  int i32x8;
typedef __attribute__((ext_vector_type(4)))  int i32x4;

union BfFrag { v16bf v; uint4 u[2]; };

static __device__ inline v8f wmma_bf16(uint4 a0, uint4 a1,
                                       uint4 b0, uint4 b1, v8f c) {
    BfFrag A, B;
    A.u[0] = a0; A.u[1] = a1;
    B.u[0] = b0; B.u[1] = b1;
    return __builtin_amdgcn_wmma_f32_16x16x32_bf16(
        false, A.v, false, B.v, (short)0, c, false, false);
}

// fp32 -> bf16 round-to-nearest-even (bit math; deterministic)
static __device__ inline unsigned short f32_to_bf16(float f) {
    unsigned int u = __float_as_uint(f);
    return (unsigned short)((u + 0x7FFFu + ((u >> 16) & 1u)) >> 16);
}
static __device__ inline unsigned int pack_bf16x2(float lo, float hi) {
    return (unsigned)f32_to_bf16(lo) | ((unsigned)f32_to_bf16(hi) << 16);
}

// ---------------------------------------------------------------------------
// Elementwise f32 -> bf16 (weights). 8 elements per thread, b128 in/out.
// ---------------------------------------------------------------------------
__global__ __launch_bounds__(256)
void cvt_f32_bf16_kernel(const float* __restrict__ src,
                         unsigned short* __restrict__ dst, int n8)
{
    int i = blockIdx.x * 256 + threadIdx.x;
    if (i >= n8) return;
    float4 f0 = ((const float4*)src)[2 * i];
    float4 f1 = ((const float4*)src)[2 * i + 1];
    uint4 u;
    u.x = pack_bf16x2(f0.x, f0.y); u.y = pack_bf16x2(f0.z, f0.w);
    u.z = pack_bf16x2(f1.x, f1.y); u.w = pack_bf16x2(f1.z, f1.w);
    ((uint4*)dst)[i] = u;
}

// ---------------------------------------------------------------------------
// x[b][512][4096] f32 -> xt[b][4096][512] bf16.  64x64 tile per block.
// Tile staged into padded LDS ([64][68] f32) by the Tensor Data Mover
// (pad_interval = 64 DW, pad_amount = 4 DW), else per-lane async-to-LDS.
// ---------------------------------------------------------------------------
__global__ __launch_bounds__(256)
void transpose_cvt_x_kernel(const float* __restrict__ x,
                            unsigned short* __restrict__ xt)
{
    __shared__ float sF[64 * 68];
    const int b   = blockIdx.z;
    const int c0  = blockIdx.y * 64;
    const int n0  = blockIdx.x * 64;
    const int tid = threadIdx.x;

#if __has_builtin(__builtin_amdgcn_tensor_load_to_lds)
    // ---- TDM path: one descriptor moves the whole 64x64 f32 tile ----
    if (tid < 32) {
        const unsigned long long ga =
            (unsigned long long)(size_t)(x + (size_t)(b * 512 + c0) * 4096 + n0);
        const unsigned ldsb = (unsigned)(size_t)&sF[0];
        u32x4 g0;
        g0[0] = 1u;                                   // count=1 (valid D#)
        g0[1] = ldsb;                                 // lds_addr (bytes)
        g0[2] = (unsigned)(ga & 0xffffffffu);         // global_addr[31:0]
        g0[3] = (unsigned)((ga >> 32) & 0x01ffffffu)  // global_addr[56:32]
              | (2u << 30);                           // type = 2 ("image")
        i32x8 g1;
        g1[0] = (2 << 16)       // data_size = 4 bytes
              | (1 << 20)       // pad_enable
              | (5 << 22)       // pad_interval: 64 DWORDs
              | (3 << 25);      // pad_amount:   4 DWORDs
        g1[1] = (int)(4096u << 16);          // tensor_dim0[15:0] (=4096)
        g1[2] = (int)(512u << 16);           // dim0[31:16]=0, tensor_dim1=512
        g1[3] = (int)(64u << 16);            // dim1[31:16]=0, tile_dim0=64
        g1[4] = (int)64u;                    // tile_dim1=64, tile_dim2=0
        g1[5] = (int)4096;                   // tensor_dim0_stride[31:0]
        g1[6] = 0;                           // stride msbs / dim1_stride
        g1[7] = 0;
        i32x4 z4 = {0, 0, 0, 0};             // groups 2/3 unused (2D tensor)
        i32x8 z8 = {0, 0, 0, 0, 0, 0, 0, 0};
        __builtin_amdgcn_tensor_load_to_lds(g0, g1, z4, z4, z8, 0);
#if __has_builtin(__builtin_amdgcn_s_wait_tensorcnt)
        __builtin_amdgcn_s_wait_tensorcnt(0);
#else
        asm volatile("s_wait_tensorcnt 0x0" ::: "memory");
#endif
    }
    __syncthreads();
#else
    // ---- fallback: per-lane async global->LDS b128 copies ----
    #pragma unroll
    for (int i = 0; i < 4; ++i) {
        int idx = tid + i * 256;
        int c   = idx >> 4;
        int g   = (idx & 15) * 4;
        const float* gp = x + (size_t)(b * 512 + c0 + c) * 4096 + n0 + g;
        unsigned lds = (unsigned)(size_t)&sF[c * 68 + g];
        asm volatile("global_load_async_to_lds_b128 %0, %1, off"
                     :: "v"(lds), "v"((unsigned long long)(size_t)gp)
                     : "memory");
    }
    asm volatile("s_wait_asynccnt 0x0" ::: "memory");
    __syncthreads();
#endif

    // transposed, converted, packed b128 stores: xt[n][c], c contiguous
    #pragma unroll
    for (int i = 0; i < 2; ++i) {
        int idx = tid + i * 256;            // 512 output 8-element groups
        int n   = idx >> 3;
        int cg  = (idx & 7) * 8;
        uint4 u;
        unsigned* w = &u.x;
        #pragma unroll
        for (int j = 0; j < 4; ++j)
            w[j] = pack_bf16x2(sF[(cg + 2*j    ) * 68 + n],
                               sF[(cg + 2*j + 1) * 68 + n]);
        *(uint4*)(xt + (size_t)(b * 4096 + n0 + n) * 512 + c0 + cg) = u;
    }
}

// ---------------------------------------------------------------------------
// LDS-free GEMM:  C[b][o][n] = sum_c At[b][n][c] * Bw[o][c] + bias[o]
// (M = n dim, N = o dim, K = c = 512). Block tile 128n x 128o, 8 waves,
// wave tile 64n x 32o. All frags are direct global b128 loads; epilogue
// stores 8 consecutive-n values packed per lane.
// ---------------------------------------------------------------------------
template<bool OUT_BF16>
__global__ __launch_bounds__(256)
void gemm_tn_kernel(const unsigned short* __restrict__ At,  // [b][4096][K]
                    const unsigned short* __restrict__ Bw,  // [ODIM][K] bf16
                    const float* __restrict__ bias,         // [ODIM]
                    void* __restrict__ Cv,                  // [b][ODIM][4096]
                    int ODIM, int K)
{
    const int b    = blockIdx.z;
    const int n0   = blockIdx.x * 128;
    const int o0   = blockIdx.y * 128;
    const int tid  = threadIdx.x;
    const int lane = tid & 31;
    const int wave = tid >> 5;
    const int wm   = (wave >> 2) * 64;
    const int wn   = (wave & 3) * 32;
    const int mrow = lane & 15;
    const int akb  = (lane >> 4) * 16;
    const int bkb  = (lane >> 4) * 16;

    const unsigned short* Ab = At + (size_t)b * 4096 * K;
    float*          Cf = (float*)Cv          + (size_t)b * ODIM * 4096;
    unsigned short* Ch = (unsigned short*)Cv + (size_t)b * ODIM * 4096;

    v8f acc[4][2];
    #pragma unroll
    for (int i = 0; i < 4; ++i)
        #pragma unroll
        for (int j = 0; j < 2; ++j) acc[i][j] = v8f{};

    const unsigned short* aB[4];
    const unsigned short* bB[2];
    #pragma unroll
    for (int mt = 0; mt < 4; ++mt)
        aB[mt] = Ab + (size_t)(n0 + wm + mt * 16 + mrow) * K + akb;
    #pragma unroll
    for (int nt = 0; nt < 2; ++nt)
        bB[nt] = Bw + (size_t)(o0 + wn + nt * 16 + mrow) * K + bkb;

    for (int k0 = 0; k0 < K; k0 += 32) {
        uint4 aF0[4], aF1[4], bF0[2], bF1[2];
        #pragma unroll
        for (int mt = 0; mt < 4; ++mt) {
            const unsigned short* p = aB[mt] + k0;
            aF0[mt] = *(const uint4*)p;
            aF1[mt] = *(const uint4*)(p + 8);
        }
        #pragma unroll
        for (int nt = 0; nt < 2; ++nt) {
            const unsigned short* p = bB[nt] + k0;
            bF0[nt] = *(const uint4*)p;
            bF1[nt] = *(const uint4*)(p + 8);
        }
        #pragma unroll
        for (int mt = 0; mt < 4; ++mt)
            #pragma unroll
            for (int nt = 0; nt < 2; ++nt)
                acc[mt][nt] = wmma_bf16(aF0[mt], aF1[mt], bF0[nt], bF1[nt],
                                        acc[mt][nt]);
    }

    const int rofs = (lane >> 4) * 8;
    #pragma unroll
    for (int mt = 0; mt < 4; ++mt)
        #pragma unroll
        for (int nt = 0; nt < 2; ++nt) {
            int o  = o0 + wn + nt * 16 + (lane & 15);
            int nb = n0 + wm + mt * 16 + rofs;
            float bz = bias[o];
            if constexpr (OUT_BF16) {
                uint4 u;
                unsigned* w = &u.x;
                #pragma unroll
                for (int j = 0; j < 4; ++j)
                    w[j] = pack_bf16x2(acc[mt][nt][2*j] + bz,
                                       acc[mt][nt][2*j + 1] + bz);
                *(uint4*)(Ch + (size_t)o * 4096 + nb) = u;
            } else {
                float4 f0, f1;
                f0.x = acc[mt][nt][0] + bz; f0.y = acc[mt][nt][1] + bz;
                f0.z = acc[mt][nt][2] + bz; f0.w = acc[mt][nt][3] + bz;
                f1.x = acc[mt][nt][4] + bz; f1.y = acc[mt][nt][5] + bz;
                f1.z = acc[mt][nt][6] + bz; f1.w = acc[mt][nt][7] + bz;
                *(float4*)(Cf + (size_t)o * 4096 + nb)     = f0;
                *(float4*)(Cf + (size_t)o * 4096 + nb + 4) = f1;
            }
        }
}

// ---------------------------------------------------------------------------
// Channel attention: one block per (b,h). q,k,v are 64x4096 bf16 rows of qkv.
// Phase 1: q.k^T (64x64, K=4096): 16 score tiles over 8 waves -> each wave
//          owns 2 whole tiles over the full K; results stored once to LDS.
// Phase 2: fp32 softmax.  Phase 3: attn.v with transposed packed output
//          Ot[b][n][512].
// ---------------------------------------------------------------------------
__global__ __launch_bounds__(256)
void attn_kernel(const unsigned short* __restrict__ qkv,
                 float* __restrict__ attn_out,          // [b][h][64][64] f32
                 unsigned short* __restrict__ Ot)       // [b][4096][512] bf16
{
    constexpr int N = 4096;
    const int bh = blockIdx.x;
    const int b  = bh >> 3;
    const int h  = bh & 7;
    const unsigned short* q = qkv + ((size_t)b * 1536 +        h * 64) * N;
    const unsigned short* k = qkv + ((size_t)b * 1536 +  512 + h * 64) * N;
    const unsigned short* v = qkv + ((size_t)b * 1536 + 1024 + h * 64) * N;

    __shared__ float          sAtt [64 * 68];
    __shared__ unsigned short sAttB[64 * 72];
    __shared__ unsigned short sVt  [128 * 72];  // [n_local][e]

    const int tid  = threadIdx.x;
    const int lane = tid & 31;
    const int wave = tid >> 5;
    const int mrow = lane & 15;
    const int akb  = (lane >> 4) * 16;
    const int bkb  = (lane >> 4) * 16;
    const int rofs = (lane >> 4) * 8;

    // ---- Phase 1: this wave owns score tiles (mt, ntp) and (mt, ntp+1) ----
    const int mt  = wave >> 1;
    const int ntp = (wave & 1) * 2;
    v8f acc[2];
    acc[0] = v8f{}; acc[1] = v8f{};

    const unsigned short* aP  = q + (size_t)(mt * 16 + mrow) * N + akb;
    const unsigned short* bP0 = k + (size_t)((ntp    ) * 16 + mrow) * N + bkb;
    const unsigned short* bP1 = k + (size_t)((ntp + 1) * 16 + mrow) * N + bkb;

    for (int kk = 0; kk < N; kk += 32) {
        uint4 a0 = *(const uint4*)(aP + kk);
        uint4 a1 = *(const uint4*)(aP + kk + 8);
        uint4 c0 = *(const uint4*)(bP0 + kk);
        uint4 c1 = *(const uint4*)(bP0 + kk + 8);
        uint4 d0 = *(const uint4*)(bP1 + kk);
        uint4 d1 = *(const uint4*)(bP1 + kk + 8);
        acc[0] = wmma_bf16(a0, a1, c0, c1, acc[0]);
        acc[1] = wmma_bf16(a0, a1, d0, d1, acc[1]);
    }
    // plain stores: every sAtt cell written exactly once
    #pragma unroll
    for (int t = 0; t < 2; ++t)
        #pragma unroll
        for (int j = 0; j < 8; ++j)
            sAtt[(mt * 16 + j + rofs) * 68 + (ntp + t) * 16 + (lane & 15)]
                = acc[t][j];
    __syncthreads();

    // ---- Phase 2: fp32 softmax per row (scale = 1/8) ----
    if (tid < 64) {
        float* r = &sAtt[tid * 68];
        float mx = -3.4e38f;
        #pragma unroll 4
        for (int e = 0; e < 64; ++e) mx = fmaxf(mx, r[e] * 0.125f);
        float sum = 0.0f;
        #pragma unroll 4
        for (int e = 0; e < 64; ++e) {
            float ex = __expf(r[e] * 0.125f - mx);
            r[e] = ex; sum += ex;
        }
        float inv = 1.0f / sum;
        float* aout = attn_out + ((size_t)bh * 64 + tid) * 64;
        #pragma unroll 4
        for (int e = 0; e < 64; ++e) {
            float a = r[e] * inv;
            aout[e] = a;
            sAttB[tid * 72 + e] = f32_to_bf16(a);
        }
    }
    __syncthreads();

    // ---- Phase 3: out = attn(64x64) . v(64x4096), n-chunks of 128 ----
    unsigned short* OtB = Ot + (size_t)b * 4096 * 512 + h * 64;
    for (int nc = 0; nc < N; nc += 128) {
        // stage v^T chunk: sVt[n_local][e], packed-pair b32 stores
        #pragma unroll
        for (int i = 0; i < 2; ++i) {
            int idx = tid + i * 256;        // 512 units: 32 e-pairs x 16 groups
            int e   = (idx >> 4) * 2;
            int g   = (idx & 15) * 8;
            uint4 u0 = *(const uint4*)(v + (size_t)e * N + nc + g);
            uint4 u1 = *(const uint4*)(v + (size_t)(e + 1) * N + nc + g);
            const unsigned* w0 = &u0.x;
            const unsigned* w1 = &u1.x;
            #pragma unroll
            for (int j = 0; j < 4; ++j) {
                unsigned lo0 = w0[j] & 0xffffu, hi0 = w0[j] >> 16;
                unsigned lo1 = w1[j] & 0xffffu, hi1 = w1[j] >> 16;
                *(unsigned*)&sVt[(g + 2*j    ) * 72 + e] = lo0 | (lo1 << 16);
                *(unsigned*)&sVt[(g + 2*j + 1) * 72 + e] = hi0 | (hi1 << 16);
            }
        }
        __syncthreads();

        const int nl = wave * 16;
        v8f oacc[4];
        #pragma unroll
        for (int m = 0; m < 4; ++m) oacc[m] = v8f{};

        #pragma unroll
        for (int ks = 0; ks < 2; ++ks) {
            const unsigned short* pb = &sVt[(nl + mrow) * 72 + ks * 32 + bkb];
            uint4 b0 = *(const uint4*)pb;
            uint4 b1 = *(const uint4*)(pb + 8);
            #pragma unroll
            for (int m = 0; m < 4; ++m) {
                const unsigned short* pa =
                    &sAttB[(m * 16 + mrow) * 72 + ks * 32 + akb];
                uint4 a0 = *(const uint4*)pa;
                uint4 a1 = *(const uint4*)(pa + 8);
                oacc[m] = wmma_bf16(a0, a1, b0, b1, oacc[m]);
            }
        }
        // transposed output: lane holds 8 consecutive d at fixed n -> b128
        #pragma unroll
        for (int m = 0; m < 4; ++m) {
            int col = nc + nl + (lane & 15);
            int d0  = m * 16 + rofs;
            uint4 u;
            unsigned* w = &u.x;
            #pragma unroll
            for (int j = 0; j < 4; ++j)
                w[j] = pack_bf16x2(oacc[m][2*j], oacc[m][2*j + 1]);
            *(uint4*)(OtB + (size_t)col * 512 + d0) = u;
        }
        __syncthreads();
    }
}

// ---------------------------------------------------------------------------
// attn.mean over heads: [b][8][64][64] -> [b][64][64]
// ---------------------------------------------------------------------------
__global__ __launch_bounds__(256)
void attn_mean_kernel(const float* __restrict__ attn, float* __restrict__ out)
{
    int i  = blockIdx.x * 256 + threadIdx.x;   // 32768 outputs
    int b  = i >> 12;
    int de = i & 4095;
    float s = 0.0f;
    #pragma unroll
    for (int h = 0; h < 8; ++h) s += attn[((size_t)(b * 8 + h) << 12) + de];
    out[i] = s * 0.125f;
}

// ---------------------------------------------------------------------------
extern "C" void kernel_launch(void* const* d_in, const int* in_sizes, int n_in,
                              void* d_out, int out_size, void* d_ws, size_t ws_size,
                              hipStream_t stream)
{
    (void)in_sizes; (void)n_in; (void)out_size; (void)ws_size;
    constexpr int B = 8, C = 512, N = 4096, M3 = 3 * C;
    const float* x      = (const float*)d_in[0];
    const float* qkv_w  = (const float*)d_in[1];
    const float* qkv_b  = (const float*)d_in[2];
    const float* proj_w = (const float*)d_in[3];
    const float* proj_b = (const float*)d_in[4];

    float* out       = (float*)d_out;                 // [8][512][64][64]
    float* attn_mean = out + (size_t)B * C * N;       // [8][64][64]

    // workspace layout (bf16 stored as ushort), all 16B aligned
    unsigned short* xt    = (unsigned short*)d_ws;            // [8][4096][512]
    unsigned short* qkvb  = xt    + (size_t)B * N * C;        // [8][1536][4096]
    unsigned short* Ot    = qkvb  + (size_t)B * M3 * N;       // [8][4096][512]
    unsigned short* qkvWb = Ot    + (size_t)B * N * C;        // [1536][512]
    unsigned short* prjWb = qkvWb + (size_t)M3 * C;           // [512][512]
    float*          attf  = (float*)(prjWb + (size_t)C * C);  // [8][8][64][64]

    // 0) convert weights to bf16
    cvt_f32_bf16_kernel<<<dim3((M3 * C / 8 + 255) / 256), 256, 0, stream>>>(
        qkv_w, qkvWb, M3 * C / 8);
    cvt_f32_bf16_kernel<<<dim3((C * C / 8 + 255) / 256), 256, 0, stream>>>(
        proj_w, prjWb, C * C / 8);

    // 1) transpose+convert x -> xt (TDM / async staging)
    transpose_cvt_x_kernel<<<dim3(N / 64, C / 64, B), 256, 0, stream>>>(x, xt);

    // 2) QKV projection (LDS-free WMMA GEMM), bf16 out row-major [o][n]
    gemm_tn_kernel<true><<<dim3(N / 128, M3 / 128, B), 256, 0, stream>>>(
        xt, qkvWb, qkv_b, qkvb, M3, C);

    // 3) channel attention per (b,h); writes transposed Ot[b][n][c]
    attn_kernel<<<dim3(B * 8), 256, 0, stream>>>(qkvb, attf, Ot);

    // 4) head-mean of attn
    attn_mean_kernel<<<dim3(128), 256, 0, stream>>>(attf, attn_mean);

    // 5) output projection (LDS-free WMMA GEMM), f32 out + bias -> d_out
    gemm_tn_kernel<false><<<dim3(N / 128, C / 128, B), 256, 0, stream>>>(
        Ot, prjWb, proj_b, out, C, C);
}
